// AttentionModel_25494925869145
// MI455X (gfx1250) — matrix operand order repaired and checked by reading, not compiled, not run on hardware
//
#include <hip/hip_runtime.h>
#include <hip/hip_bf16.h>
#include <math.h>

// MI455X / gfx1250, wave32. bf16 WMMA for GEMMs, bf16 storage of K/V/logit_K
// to halve per-step HBM traffic (bandwidth-bound decode loop).

typedef __attribute__((ext_vector_type(16))) __bf16 v16bf;
typedef __attribute__((ext_vector_type(8)))  __bf16 v8bf;
typedef __attribute__((ext_vector_type(8)))  float  v8f;

#define B_   256
#define N_   1000
#define D_   128
#define H_   8
#define DK_  16
#define T_   100
#define NEG_INF_ (-1e9f)

__device__ __forceinline__ float warp_max(float v) {
  #pragma unroll
  for (int o = 16; o; o >>= 1) v = fmaxf(v, __shfl_xor(v, o, 32));
  return v;
}
__device__ __forceinline__ float warp_sum(float v) {
  #pragma unroll
  for (int o = 16; o; o >>= 1) v += __shfl_xor(v, o, 32);
  return v;
}

// ---------- init ----------
__global__ void k_zero_mask(int* __restrict__ mask) {
  int i = blockIdx.x * blockDim.x + threadIdx.x;
  if (i < B_ * N_) mask[i] = 0;
}

// ---------- pack a row-major f32 weight into the WMMA B-operand lane layout ----------
// B tile (32x16 bf16): lane half h, elem e holds K=8h+e (e<8), K=16+8h+(e-8) (e>=8);
// column N = lane&15. Blocks stored contiguously: bi = kk*ntilesN + nt.
__global__ void k_layoutB(const float* __restrict__ W, __bf16* __restrict__ dst,
                          int ncols, int ntilesN) {
  int id = blockIdx.x * blockDim.x + threadIdx.x;
  int total = 4 * ntilesN * 512;
  if (id >= total) return;
  int bi = id >> 9;
  int within = id & 511;
  int l = within >> 4;
  int e = within & 15;
  int kk = bi / ntilesN;
  int nt = bi % ntilesN;
  int kg = kk * 32 + ((e & 8) << 1) + (l >> 4) * 8 + (e & 7);
  int ng = nt * 16 + (l & 15);
  dst[id] = (__bf16)W[kg * ncols + ng];
}

// ---------- precompute GEMM: kvl = emb[B*N,128] @ W_node[128,384] -> bf16 K/V/LK ----------
// one wave per 16(M) x 64(N) tile; A converted f32->bf16 in-register.
__global__ void k_kvl_gemm(const float* __restrict__ emb, const __bf16* __restrict__ wlay,
                           __bf16* __restrict__ Kb, __bf16* __restrict__ Vb,
                           __bf16* __restrict__ LKb) {
  const int lane = threadIdx.x;            // 0..31
  const int m0 = blockIdx.x * 16;
  const int row = m0 + (lane & 15);
  const int hi_half = lane >> 4;           // 0 or 1
  const int kbase = hi_half ? 8 : 0;
  v8f c[4];
  #pragma unroll
  for (int j = 0; j < 4; j++)
    #pragma unroll
    for (int i = 0; i < 8; i++) c[j][i] = 0.f;

  #pragma unroll
  for (int kk = 0; kk < 4; kk++) {
    const float* ap = emb + (size_t)row * D_ + kk * 32 + kbase;
    v16bf a;
    #pragma unroll
    for (int i = 0; i < 8; i++) a[i] = (__bf16)ap[i];        // K = kk*32+kbase+0..7
    #pragma unroll
    for (int i = 0; i < 8; i++) a[8 + i] = (__bf16)ap[16 + i]; // K = +16..23
    #pragma unroll
    for (int j = 0; j < 4; j++) {
      int bi = kk * 24 + (blockIdx.y * 4 + j);
      const v16bf b = *(const v16bf*)(wlay + (size_t)bi * 512 + lane * 16);
      c[j] = __builtin_amdgcn_wmma_f32_16x16x32_bf16(false, a, false, b,
                                                     (short)0, c[j], false, false);
    }
  }
  const int nloc = lane & 15;
  const int mofs = hi_half ? 8 : 0;
  #pragma unroll
  for (int j = 0; j < 4; j++) {
    int ng = blockIdx.y * 64 + j * 16 + nloc;   // 0..383
    int sec = ng >> 7;                           // 0:K 1:V 2:logit_K
    int col = ng & 127;
    __bf16* dst = (sec == 0) ? Kb : ((sec == 1) ? Vb : LKb);
    #pragma unroll
    for (int r = 0; r < 8; r++) {
      int m = m0 + r + mofs;
      dst[(size_t)m * D_ + col] = (__bf16)c[j][r];
    }
  }
}

// ---------- mean over N and fixed context ----------
__global__ void k_mean(const float* __restrict__ emb, float* __restrict__ meanv) {
  int b = blockIdx.x, d = threadIdx.x;
  float s = 0.f;
  for (int n = 0; n < N_; n++) s += emb[((size_t)b * N_ + n) * D_ + d];
  meanv[b * D_ + d] = s * (1.f / N_);
}
__global__ void k_fixed(const float* __restrict__ meanv, const float* __restrict__ Wfixed,
                        float* __restrict__ fixedc) {
  __shared__ float mrow[D_];
  int b = blockIdx.x, d = threadIdx.x;
  mrow[d] = meanv[b * D_ + d];
  __syncthreads();
  float s = 0.f;
  for (int k = 0; k < D_; k++) s += mrow[k] * Wfixed[k * D_ + d];
  fixedc[b * D_ + d] = s;
}

// ---------- per-step: query = fixed_ctx + last_emb @ W_step ----------
__global__ void k_query(const float* __restrict__ emb, const float* __restrict__ fixedc,
                        const float* __restrict__ Wstep, const int* __restrict__ sel,
                        float* __restrict__ query, int t) {
  __shared__ float last[D_];
  int b = blockIdx.x, d = threadIdx.x;
  if (t == 0) {
    last[d] = 0.f;
  } else {
    int s = sel[b];
    last[d] = emb[((size_t)b * N_ + s) * D_ + d];
  }
  __syncthreads();
  float acc = fixedc[b * D_ + d];
  for (int k = 0; k < D_; k++) acc += last[k] * Wstep[k * D_ + d];
  query[b * D_ + d] = acc;
}

// ---------- per-step: masked glimpse attention -> heads (bf16) ----------
__global__ void k_attn(const float* __restrict__ query, const __bf16* __restrict__ Kb,
                       const __bf16* __restrict__ Vb, const int* __restrict__ mask,
                       __bf16* __restrict__ headsb) {
  __shared__ float qs[D_];
  __shared__ float compat[H_ * N_];  // 32 KB
  const int b = blockIdx.x, tid = threadIdx.x;
  if (tid < D_) qs[tid] = query[b * D_ + tid];
  __syncthreads();
  const float inv_sqrt_dk = 0.25f;  // 1/sqrt(16)
  for (int idx = tid; idx < H_ * N_; idx += blockDim.x) {
    int n = idx >> 3, h = idx & 7;  // 8 consecutive tids cover one 256B K row
    const __bf16* kp = Kb + ((size_t)b * N_ + n) * D_ + h * DK_;
    float s = 0.f;
    #pragma unroll
    for (int d = 0; d < DK_; d++) s += qs[h * DK_ + d] * (float)kp[d];
    compat[h * N_ + n] = mask[b * N_ + n] ? NEG_INF_ : s * inv_sqrt_dk;
  }
  __syncthreads();
  const int h = tid >> 5, lane = tid & 31;  // one wave per head
  float m = -INFINITY;
  for (int n = lane; n < N_; n += 32) m = fmaxf(m, compat[h * N_ + n]);
  m = warp_max(m);
  float acc[DK_];
  #pragma unroll
  for (int d = 0; d < DK_; d++) acc[d] = 0.f;
  float ssum = 0.f;
  for (int n = lane; n < N_; n += 32) {
    float e = __expf(compat[h * N_ + n] - m);
    ssum += e;
    const __bf16* vp = Vb + ((size_t)b * N_ + n) * D_ + h * DK_;
    #pragma unroll
    for (int d = 0; d < DK_; d++) acc[d] += e * (float)vp[d];
  }
  ssum = warp_sum(ssum);
  #pragma unroll
  for (int d = 0; d < DK_; d++) acc[d] = warp_sum(acc[d]);
  if (lane == 0) {
    float inv = 1.f / ssum;
    for (int d = 0; d < DK_; d++)
      headsb[b * D_ + h * DK_ + d] = (__bf16)(acc[d] * inv);
  }
}

// ---------- per-step WMMA GEMM: glimpse = heads[256,128] @ W_out[128,128] ----------
__global__ void k_glimpse(const __bf16* __restrict__ heads, const __bf16* __restrict__ wlay,
                          float* __restrict__ glimpse) {
  const int lane = threadIdx.x;
  const int m0 = blockIdx.x * 16;
  const int n0 = blockIdx.y * 16;
  const int row = m0 + (lane & 15);
  const int hi_half = lane >> 4;
  const int kbase = hi_half ? 8 : 0;
  v8f c;
  #pragma unroll
  for (int i = 0; i < 8; i++) c[i] = 0.f;
  #pragma unroll
  for (int kk = 0; kk < 4; kk++) {
    const __bf16* ap = heads + (size_t)row * D_ + kk * 32 + kbase;
    v8bf lo = *(const v8bf*)ap;
    v8bf hi = *(const v8bf*)(ap + 16);
    v16bf a;
    #pragma unroll
    for (int i = 0; i < 8; i++) { a[i] = lo[i]; a[8 + i] = hi[i]; }
    int bi = kk * 8 + blockIdx.y;
    const v16bf b = *(const v16bf*)(wlay + (size_t)bi * 512 + lane * 16);
    c = __builtin_amdgcn_wmma_f32_16x16x32_bf16(false, a, false, b,
                                                (short)0, c, false, false);
  }
  const int nloc = lane & 15;
  const int mofs = hi_half ? 8 : 0;
  #pragma unroll
  for (int r = 0; r < 8; r++)
    glimpse[(size_t)(m0 + r + mofs) * D_ + n0 + nloc] = c[r];
}

// ---------- per-step: logits, log_softmax, greedy argmax, mask update ----------
__global__ void k_logits(const float* __restrict__ glimpse, const __bf16* __restrict__ LKb,
                         int* __restrict__ mask, int* __restrict__ sel,
                         float* __restrict__ logp_out, float* __restrict__ pi_out, int t) {
  __shared__ float g[D_];
  __shared__ float lg[N_];
  __shared__ float red[256];
  __shared__ int redi[256];
  const int b = blockIdx.x, tid = threadIdx.x;
  if (tid < D_) g[tid] = glimpse[b * D_ + tid];
  __syncthreads();
  const float inv_sqrt_d = 0.08838834764831845f;  // 1/sqrt(128)
  for (int n = tid; n < N_; n += blockDim.x) {
    const __bf16* kp = LKb + ((size_t)b * N_ + n) * D_;
    float s = 0.f;
    for (int d = 0; d < D_; d++) s += g[d] * (float)kp[d];
    s = 10.f * tanhf(s * inv_sqrt_d);
    lg[n] = mask[b * N_ + n] ? NEG_INF_ : s;
  }
  __syncthreads();
  // max
  float mx = -INFINITY;
  for (int n = tid; n < N_; n += blockDim.x) mx = fmaxf(mx, lg[n]);
  red[tid] = mx;
  __syncthreads();
  for (int s = 128; s; s >>= 1) {
    if (tid < s) red[tid] = fmaxf(red[tid], red[tid + s]);
    __syncthreads();
  }
  mx = red[0];
  __syncthreads();
  // logsumexp
  float se = 0.f;
  for (int n = tid; n < N_; n += blockDim.x) se += __expf(lg[n] - mx);
  red[tid] = se;
  __syncthreads();
  for (int s = 128; s; s >>= 1) {
    if (tid < s) red[tid] += red[tid + s];
    __syncthreads();
  }
  float logZ = mx + __logf(red[0]);
  __syncthreads();
  // write log_p and find first argmax
  float bv = -INFINITY;
  int bidx = N_;
  for (int n = tid; n < N_; n += blockDim.x) {
    float v = lg[n];
    logp_out[(size_t)t * B_ * N_ + (size_t)b * N_ + n] = v - logZ;
    if (v > bv) { bv = v; bidx = n; }  // ascending n: strict '>' keeps first
  }
  red[tid] = bv;
  redi[tid] = bidx;
  __syncthreads();
  for (int s = 128; s; s >>= 1) {
    if (tid < s) {
      if (red[tid + s] > red[tid] ||
          (red[tid + s] == red[tid] && redi[tid + s] < redi[tid])) {
        red[tid] = red[tid + s];
        redi[tid] = redi[tid + s];
      }
    }
    __syncthreads();
  }
  if (tid == 0) {
    int s0 = redi[0];
    sel[b] = s0;
    mask[b * N_ + s0] = 1;
    pi_out[(size_t)t * B_ + b] = (float)s0;
  }
}

extern "C" void kernel_launch(void* const* d_in, const int* in_sizes, int n_in,
                              void* d_out, int out_size, void* d_ws, size_t ws_size,
                              hipStream_t stream) {
  (void)in_sizes; (void)n_in; (void)out_size; (void)ws_size;
  const float* emb    = (const float*)d_in[0];  // [256,1000,128]
  const float* Wnode  = (const float*)d_in[1];  // [128,384]
  const float* Wfixed = (const float*)d_in[2];  // [128,128]
  const float* Wstep  = (const float*)d_in[3];  // [128,128]
  const float* Wout   = (const float*)d_in[4];  // [128,128]
  // d_in[5] = T; hardcoded T_=100 (cannot read device scalar under graph capture)

  char* ws = (char*)d_ws;
  size_t o = 0;
  auto take = [&](size_t bytes) -> char* {
    char* p = ws + o;
    o = (o + bytes + 255) & ~(size_t)255;
    return p;
  };
  const size_t BND = (size_t)B_ * N_ * D_;          // 32,768,000
  __bf16* Kb        = (__bf16*)take(BND * 2);
  __bf16* Vb        = (__bf16*)take(BND * 2);
  __bf16* LKb       = (__bf16*)take(BND * 2);
  __bf16* wnode_lay = (__bf16*)take((size_t)4 * 24 * 512 * 2);
  __bf16* wout_lay  = (__bf16*)take((size_t)4 * 8 * 512 * 2);
  float*  meanv     = (float*)take((size_t)B_ * D_ * 4);
  float*  fixedc    = (float*)take((size_t)B_ * D_ * 4);
  float*  query     = (float*)take((size_t)B_ * D_ * 4);
  __bf16* headsb    = (__bf16*)take((size_t)B_ * D_ * 2);
  float*  glimpse   = (float*)take((size_t)B_ * D_ * 4);
  int*    mask      = (int*)take((size_t)B_ * N_ * 4);
  int*    sel       = (int*)take((size_t)B_ * 4);

  float* logp_out = (float*)d_out;                         // [T,B,N]
  float* pi_out   = (float*)d_out + (size_t)T_ * B_ * N_;  // [T,B]

  // init + precompute
  k_zero_mask<<<(B_ * N_ + 255) / 256, 256, 0, stream>>>(mask);
  k_layoutB<<<(4 * 24 * 512 + 255) / 256, 256, 0, stream>>>(Wnode, wnode_lay, 384, 24);
  k_layoutB<<<(4 * 8 * 512 + 255) / 256, 256, 0, stream>>>(Wout, wout_lay, 128, 8);
  k_kvl_gemm<<<dim3(B_ * N_ / 16, 6), 32, 0, stream>>>(emb, wnode_lay, Kb, Vb, LKb);
  k_mean<<<B_, D_, 0, stream>>>(emb, meanv);
  k_fixed<<<B_, D_, 0, stream>>>(meanv, Wfixed, fixedc);

  // sequential greedy decode
  for (int t = 0; t < T_; t++) {
    k_query<<<B_, D_, 0, stream>>>(emb, fixedc, Wstep, sel, query, t);
    k_attn<<<B_, 256, 0, stream>>>(query, Kb, Vb, mask, headsb);
    k_glimpse<<<dim3(B_ / 16, D_ / 16), 32, 0, stream>>>(headsb, wout_lay, glimpse);
    k_logits<<<B_, 256, 0, stream>>>(glimpse, LKb, mask, sel, logp_out, pi_out, t);
  }
}
